// Policy_6794638263058
// MI455X (gfx1250) — matrix operand order repaired and checked by reading, not compiled, not run on hardware
//
#include <hip/hip_runtime.h>
#include <hip/hip_bf16.h>

#define N_QUBITS 12
#define DIM 4096
#define BATCH 4096
#define LAYERS 3
#define CHUNK 512   // batch rows per sim->gemm launch pair; qbuf = CHUNK*DIM*2B = 4 MB

typedef __attribute__((ext_vector_type(16))) __bf16 v16bf;
typedef __attribute__((ext_vector_type(8)))  float  v8f;

__device__ __forceinline__ unsigned short f2bf(float f) {
  // round-to-nearest-even f32 -> bf16 (payloads here are finite)
  unsigned int x = __float_as_uint(f);
  unsigned int r = x + 0x7fffu + ((x >> 16) & 1u);
  return (unsigned short)(r >> 16);
}

// ---------------------------------------------------------------------------
// Kernel 0: convert Wq (14 x DIM f32) -> zero-padded (16 x DIM) bf16, once.
// Output as packed dwords (2 bf16 per uint).
// ---------------------------------------------------------------------------
__global__ __launch_bounds__(256)
void wq_to_bf16_kernel(const float* __restrict__ Wq, unsigned int* __restrict__ wqbf)
{
  const int row = blockIdx.x;            // 0..15 (14,15 are zero pad)
  const int t   = threadIdx.x;
  #pragma unroll
  for (int r = 0; r < (DIM / 2) / 256; ++r) {
    int dw = t + 256 * r;                // dword index within row, 0..2047
    unsigned int pk = 0;
    if (row < 14) {
      const float* wp = Wq + (size_t)row * DIM + 2 * dw;
      pk = (unsigned int)f2bf(wp[0]) | ((unsigned int)f2bf(wp[1]) << 16);
    }
    wqbf[(size_t)row * (DIM / 2) + dw] = pk;
  }
}

// ---------------------------------------------------------------------------
// Kernel 1: full 12-qubit circuit for ONE batch element per workgroup,
// state resident in LDS (32 KB). Writes q = |amp|^2 as bf16 to scratch.
// ---------------------------------------------------------------------------
__global__ __launch_bounds__(256)
void qsim_kernel(const float* __restrict__ x,    // (BATCH,14)
                 const float* __restrict__ w1,   // (504,)
                 const float* __restrict__ t1,   // (504,)
                 unsigned short* __restrict__ qout, // bf16, (CHUNK, DIM)
                 int batch_base)
{
  __shared__ float sre[DIM];
  __shared__ float sim_[DIM];
  __shared__ float xs[14];
  const int t = threadIdx.x;
  const int b = batch_base + blockIdx.x;

  if (t < 14) xs[t] = x[b * 14 + t];
  #pragma unroll
  for (int r = 0; r < DIM / 256; ++r) { int d = t + 256 * r; sre[d] = 0.f; sim_[d] = 0.f; }
  if (t == 0) sre[0] = 1.0f;               // |0...0>
  __syncthreads();

  const int kindY[14] = {0,1,0,0,1,0,0,1,0,0,1,0,0,1}; // z,y,z pattern

  for (int k = 0; k < LAYERS; ++k) {
    for (int i = 0; i < N_QUBITS; ++i) {
      // ---- fuse 14 rotations into one 2x2 complex matrix (redundant per thread,
      // uniform addresses -> scalar loads; trivially cheap) ----
      float m00r = 1.f, m00i = 0.f, m01r = 0.f, m01i = 0.f;
      float m10r = 0.f, m10i = 0.f, m11r = 1.f, m11i = 0.f;
      const int base = 14 * i + 14 * N_QUBITS * k;
      #pragma unroll
      for (int j = 0; j < 14; ++j) {
        float h = 0.5f * (w1[base + j] * xs[j] + t1[base + j]);
        float s, c;
        __sincosf(h, &s, &c);
        if (kindY[j]) {        // ry: M = [[c,-s],[s,c]] @ M  (real c,s)
          float ar, ai;
          ar = c * m00r - s * m10r; ai = c * m00i - s * m10i;
          m10r = s * m00r + c * m10r; m10i = s * m00i + c * m10i;
          m00r = ar; m00i = ai;
          ar = c * m01r - s * m11r; ai = c * m01i - s * m11i;
          m11r = s * m01r + c * m11r; m11i = s * m01i + c * m11i;
          m01r = ar; m01i = ai;
        } else {               // rz: row0 *= e^{-ih} = (c,-s); row1 *= e^{+ih} = (c,s)
          float ar, ai;
          ar = c * m00r + s * m00i; ai = c * m00i - s * m00r; m00r = ar; m00i = ai;
          ar = c * m01r + s * m01i; ai = c * m01i - s * m01r; m01r = ar; m01i = ai;
          ar = c * m10r - s * m10i; ai = c * m10i + s * m10r; m10r = ar; m10i = ai;
          ar = c * m11r - s * m11i; ai = c * m11i + s * m11r; m11r = ar; m11i = ai;
        }
      }
      // ---- apply to qubit i: pairs at stride 2^(11-i); each thread owns whole pairs ----
      const int sl = 11 - i;
      const int stride = 1 << sl;
      #pragma unroll
      for (int r = 0; r < (DIM / 2) / 256; ++r) {
        int p  = t + 256 * r;
        int i0 = ((p >> sl) << (sl + 1)) | (p & (stride - 1));
        int i1 = i0 + stride;
        float a0r = sre[i0], a0i = sim_[i0], a1r = sre[i1], a1i = sim_[i1];
        sre[i0]  = m00r * a0r - m00i * a0i + m01r * a1r - m01i * a1i;
        sim_[i0] = m00r * a0i + m00i * a0r + m01r * a1i + m01i * a1r;
        sre[i1]  = m10r * a0r - m10i * a0i + m11r * a1r - m11i * a1i;
        sim_[i1] = m10r * a0i + m10i * a0r + m11r * a1i + m11i * a1r;
      }
      __syncthreads();
    }
    if (k != LAYERS - 1) {   // entangling ring: Hcz(j,j+1) for j<11, then Hcz(11,0)
      const float rs2 = 0.70710678118654752440f;
      for (int e = 0; e < N_QUBITS; ++e) {
        const int ctrl = (e < N_QUBITS - 1) ? e : N_QUBITS - 1;
        const int tgt  = (e < N_QUBITS - 1) ? e + 1 : 0;
        const int sl = 11 - tgt;
        const int stride = 1 << sl;
        const int cb = 11 - ctrl;
        #pragma unroll
        for (int r = 0; r < (DIM / 2) / 256; ++r) {
          int p  = t + 256 * r;
          int i0 = ((p >> sl) << (sl + 1)) | (p & (stride - 1));
          int i1 = i0 + stride;
          float a0r = sre[i0], a0i = sim_[i0], a1r = sre[i1], a1i = sim_[i1];
          float n0r = (a0r + a1r) * rs2, n0i = (a0i + a1i) * rs2;
          float n1r = (a0r - a1r) * rs2, n1i = (a0i - a1i) * rs2;
          float sgn = ((i0 >> cb) & 1) ? -1.0f : 1.0f;  // CZ sign on tgt=1 & ctrl=1
          sre[i0] = n0r;       sim_[i0] = n0i;
          sre[i1] = sgn * n1r; sim_[i1] = sgn * n1i;
        }
        __syncthreads();
      }
    }
  }
  // q = |amp|^2 -> bf16, coalesced store
  #pragma unroll
  for (int r = 0; r < DIM / 256; ++r) {
    int d = t + 256 * r;
    float q = sre[d] * sre[d] + sim_[d] * sim_[d];
    qout[(size_t)blockIdx.x * DIM + d] = f2bf(q);
  }
}

// ---------------------------------------------------------------------------
// Kernel 2: logits = Q @ Wq^T + bq, softmax(14).  One wave per 16-row tile.
// bf16 WMMA, f32 accumulate, K = 4096 in steps of 64 (2 independent chains).
// ---------------------------------------------------------------------------
__global__ __launch_bounds__(32)
void proj_softmax_kernel(const unsigned int* __restrict__ q32,  // bf16 pairs, (CHUNK, DIM)
                         const unsigned int* __restrict__ w32,  // bf16 pairs, (16, DIM), pre-padded
                         const float* __restrict__ bq,          // (14,)
                         float* __restrict__ out,               // (BATCH, 14)
                         int batch_base)
{
  const int lane  = threadIdx.x;
  const int half  = lane >> 4;
  const int idx16 = lane & 15;
  const int rowTile = blockIdx.x * 16;       // chunk-local
  const int arow = rowTile + idx16;          // A-matrix row (M = lane%16)
  const int col  = idx16;                    // B-matrix column = class index

  const unsigned int* aBase = q32 + (size_t)arow * (DIM / 2);
  const unsigned int* bBase = w32 + (size_t)col  * (DIM / 2) + 8 * half; // 16*half bf16
  union { v16bf v; unsigned int d[8]; } A0, B0, A1, B1;
  v8f acc0 = {}, acc1 = {};

  for (int kk = 0; kk < DIM; kk += 64) {
    // A 16x32 bf16, k order used here (matches B): VGPR u holds K pair kOf = 2*u + 16*half
    #pragma unroll
    for (int u = 0; u < 8; ++u) {
      int kOf = 2 * u + 16 * half;  // even
      A0.d[u] = aBase[(kk + kOf) >> 1];
      A1.d[u] = aBase[(kk + 32 + kOf) >> 1];
    }
    // B 32x16 bf16, k order used here (matches A): lane = col, lanes 0-15 hold K=0..15, 16-31 hold K=16..31;
    // VGPR u holds K = 16*half + 2u, +1 -> 8 consecutive dwords (merges to b128 loads)
    #pragma unroll
    for (int u = 0; u < 8; ++u) {
      B0.d[u] = bBase[(kk >> 1) + u];
      B1.d[u] = bBase[((kk + 32) >> 1) + u];
    }
    acc0 = __builtin_amdgcn_wmma_f32_16x16x32_bf16(false, A0.v, false, B0.v,
                                                   (short)0, acc0, false, false);
    acc1 = __builtin_amdgcn_wmma_f32_16x16x32_bf16(false, A1.v, false, B1.v,
                                                   (short)0, acc1, false, false);
  }
  v8f acc = acc0 + acc1;

  // D layout: element (m,n) -> lane = n + 16*(m/8), vgpr r = m%8. Scatter to LDS.
  __shared__ float ld[16][17];
  #pragma unroll
  for (int r = 0; r < 8; ++r) ld[r + 8 * half][idx16] = acc[r];
  __syncthreads();

  if (lane < 16) {
    const int m = lane;
    float v[14];
    float mx = -1e30f;
    #pragma unroll
    for (int n = 0; n < 14; ++n) { v[n] = ld[m][n] + bq[n]; mx = fmaxf(mx, v[n]); }
    float s = 0.f;
    #pragma unroll
    for (int n = 0; n < 14; ++n) { v[n] = __expf(v[n] - mx); s += v[n]; }
    float inv = 1.0f / s;
    const int grow = batch_base + rowTile + m;
    #pragma unroll
    for (int n = 0; n < 14; ++n) out[grow * 14 + n] = v[n] * inv;
  }
}

// ---------------------------------------------------------------------------
extern "C" void kernel_launch(void* const* d_in, const int* in_sizes, int n_in,
                              void* d_out, int out_size, void* d_ws, size_t ws_size,
                              hipStream_t stream) {
  (void)in_sizes; (void)n_in; (void)out_size; (void)ws_size;
  const float* x  = (const float*)d_in[0];
  const float* w1 = (const float*)d_in[1];
  const float* t1 = (const float*)d_in[2];
  const float* Wq = (const float*)d_in[3];
  const float* bq = (const float*)d_in[4];
  float* out = (float*)d_out;

  unsigned short* qbuf = (unsigned short*)d_ws;          // CHUNK*DIM bf16 = 4 MB
  unsigned int*   wqbf = (unsigned int*)((char*)d_ws + (size_t)CHUNK * DIM * 2); // 128 KB

  wq_to_bf16_kernel<<<16, 256, 0, stream>>>(Wq, wqbf);
  for (int base = 0; base < BATCH; base += CHUNK) {
    qsim_kernel<<<CHUNK, 256, 0, stream>>>(x, w1, t1, qbuf, base);
    proj_softmax_kernel<<<CHUNK / 16, 32, 0, stream>>>(
        (const unsigned int*)qbuf, wqbf, bq, out, base);
  }
}
